// StateGNNEncoder_39702677684855
// MI455X (gfx1250) — matrix operand order repaired and checked by reading, not compiled, not run on hardware
//
#include <hip/hip_runtime.h>

typedef __attribute__((ext_vector_type(16))) __bf16 v16bf;
typedef __attribute__((ext_vector_type(8)))  __bf16 v8bf;
typedef __attribute__((ext_vector_type(8)))  float  v8f;

#define NG_  50000
#define NS_  50000
#define DG_  128
#define H_   256
#define OUT_ 64

// gfx1250 async global->LDS path, probe-gated so the file compiles on any toolchain.
#if defined(__AMDGCN__) && __has_builtin(__builtin_amdgcn_global_load_async_to_lds_b128) && __has_builtin(__builtin_amdgcn_s_wait_asynccnt)
#define USE_ASYNC_LDS 1
typedef int gv4i __attribute__((vector_size(16)));   // matches builtin's param type
#else
#define USE_ASYNC_LDS 0
#endif

__device__ __forceinline__ __bf16 f2bf(float f) {
  // round-to-nearest-even f32 -> bf16
  unsigned u = __builtin_bit_cast(unsigned, f);
  unsigned r = u + 0x7FFFu + ((u >> 16) & 1u);
  unsigned short h = (unsigned short)(r >> 16);
  return __builtin_bit_cast(__bf16, h);
}

__device__ __forceinline__ float bf2f(__bf16 b) {
  unsigned short s = __builtin_bit_cast(unsigned short, b);
  unsigned u = (unsigned)s << 16;
  return __builtin_bit_cast(float, u);
}

__global__ void zero_f32(float* __restrict__ p, long n) {
  long i = (long)blockIdx.x * blockDim.x + threadIdx.x;
  long stride = (long)gridDim.x * blockDim.x;
  for (; i < n; i += stride) p[i] = 0.0f;
}

__global__ void cvt_f32_bf16(__bf16* __restrict__ dst, const float* __restrict__ src,
                             long n) {
  long i = (long)blockIdx.x * blockDim.x + threadIdx.x;
  long stride = (long)gridDim.x * blockDim.x;
  for (; i < n; i += stride) dst[i] = f2bf(src[i]);
}

// Per-edge scatter-add: gather 8 bf16 (16B) per thread from src row, f32-atomic
// into agg[dst]; degree into cnt[dst]. Feature matrices are L2-resident.
__global__ void scatter_add_bf(const __bf16* __restrict__ feat,
                               const int* __restrict__ ei,
                               float* __restrict__ agg, float* __restrict__ cnt,
                               int E, int d) {
  int cpe = d >> 3; // 8-element chunks per edge
  long tid = (long)blockIdx.x * blockDim.x + threadIdx.x;
  long total = (long)E * cpe;
  if (tid >= total) return;
  int e = (int)(tid / cpe);
  int c = (int)(tid - (long)e * cpe);
  int s = ei[e];       // row 0: src
  int t = ei[E + e];   // row 1: dst
  v8bf v = *(const v8bf*)(feat + (long)s * d + c * 8);
  float* a = agg + (long)t * d + c * 8;
#pragma unroll
  for (int j = 0; j < 8; ++j) atomicAdd(a + j, bf2f(v[j]));
  if (c == 0) atomicAdd(cnt + t, 1.0f);
}

// Build fused bf16 A operand: Ap[m, 0:d] = agg[m,:]/max(cnt,1); Ap[m, d:2d] = xdst[m,:]
__global__ void pack_a(__bf16* __restrict__ Ap, const float* __restrict__ agg,
                       const float* __restrict__ cnt, const __bf16* __restrict__ xdst,
                       int M, int d) {
  int K = 2 * d;
  long i = (long)blockIdx.x * blockDim.x + threadIdx.x;
  long total = (long)M * K;
  if (i >= total) return;
  int m = (int)(i / K);
  int k = (int)(i - (long)m * K);
  __bf16 v;
  if (k < d) {
    float r = 1.0f / fmaxf(cnt[m], 1.0f);
    v = f2bf(agg[(long)m * d + k] * r);
  } else {
    v = xdst[(long)m * d + (k - d)];
  }
  Ap[i] = v;
}

// Build transposed bf16 weights: dst[n*K + k] = (k<K0 ? Wl[k][n] : Wr[k-K0][n])
__global__ void wprep(__bf16* __restrict__ dst, const float* __restrict__ Wl,
                      const float* __restrict__ Wr, int K0, int K1, int Nn) {
  int K = K0 + K1;
  int i = blockIdx.x * blockDim.x + threadIdx.x;
  if (i >= Nn * K) return;
  int n = i / K;
  int k = i - n * K;
  float v = (k < K0) ? Wl[(long)k * Nn + n] : Wr[(long)(k - K0) * Nn + n];
  dst[i] = f2bf(v);
}

// Pure-bf16 WMMA GEMM: C[M,Nn] = act(A[M,K] @ Bt[Nn,K]^T + bias)
// Block = 8 waves x 16 rows = 128 rows, all sharing one 64-column N supertile.
// The 64 x K B tile is staged in LDS (async global->LDS when available), then
// each wave runs a fully unrolled K loop: A from global (L2), B via ds_load.
template <int K, bool RELU, typename OutT>
__global__ void __launch_bounds__(256)
wmma_gemm(const __bf16* __restrict__ A,   // [M, K] row-major bf16
          const __bf16* __restrict__ Bt,  // [Nn, K] row-major bf16 (B transposed)
          const float* __restrict__ bias, // [Nn]
          OutT* __restrict__ C,           // [M, Nn]
          int M, int Nn) {
  constexpr int LDSK = K + 8;             // padded row stride: conflict-free banks
  __shared__ __bf16 ldsb[64 * LDSK];

  const int lane = threadIdx.x & 31;
  const int wave = threadIdx.x >> 5;
  const int lo = lane & 15;
  const int h  = lane >> 4;
  const int n0 = blockIdx.y << 6;

  // ---- cooperative B-tile fill: 64 rows x K bf16 (all 256 threads) ----
  {
    constexpr int CH_PER_ROW = K / 8;           // 16B chunks per row
    constexpr int TOT_CH = 64 * CH_PER_ROW;
    for (int q = threadIdx.x; q < TOT_CH; q += 256) {
      int c = q / CH_PER_ROW;
      int x = q - c * CH_PER_ROW;
      const __bf16* gp = Bt + (long)(n0 + c) * K + x * 8;
      __bf16* lp = &ldsb[c * LDSK + x * 8];
#if USE_ASYNC_LDS
      __builtin_amdgcn_global_load_async_to_lds_b128((gv4i*)gp, (gv4i*)lp, 0, 0);
#else
      *(v8bf*)lp = *(const v8bf*)gp;
#endif
    }
#if USE_ASYNC_LDS
    __builtin_amdgcn_s_wait_asynccnt(0);
#endif
    __syncthreads();
  }

  // ---- per-wave 16x64 tile ----
  const int mt = blockIdx.x * 8 + wave;          // wave-uniform m-tile id
  const bool valid = (mt < (M >> 4));            // wave-uniform
  if (valid) {
    const int m0 = mt << 4;
    const __bf16* arow = A + (long)(m0 + lo) * K;

    v8f acc[4] = {};

#pragma unroll
    for (int k0 = 0; k0 < K; k0 += 32) {
      // A fragment (16-bit A 16x32 layout): j=0..7 -> K=k0+8h.. ; j=8..15 -> k0+16+8h..
      v8bf a0 = *(const v8bf*)(arow + k0 + (h << 3));
      v8bf a1 = *(const v8bf*)(arow + k0 + 16 + (h << 3));
      v16bf a;
#pragma unroll
      for (int j = 0; j < 8; ++j) { a[j] = a0[j]; a[8 + j] = a1[j]; }

      // All 4 B fragments first (ds_load_b128 pairs), then 4 WMMAs.
      v16bf bf[4];
#pragma unroll
      for (int t = 0; t < 4; ++t) {
        const __bf16* bp = &ldsb[(16 * t + lo) * LDSK + k0 + (h << 3)];
        v8bf b0 = *(const v8bf*)(bp);
        v8bf b1 = *(const v8bf*)(bp + 16);
#pragma unroll
        for (int j = 0; j < 8; ++j) { bf[t][j] = b0[j]; bf[t][8 + j] = b1[j]; }
      }
#pragma unroll
      for (int t = 0; t < 4; ++t)
        acc[t] = __builtin_amdgcn_wmma_f32_16x16x32_bf16(false, a, false, bf[t],
                                                         (short)0, acc[t], false, false);
    }

    // C/D layout: element r of acc[t] -> M = m0 + r + 8h, N = n0 + 16t + lo
#pragma unroll
    for (int t = 0; t < 4; ++t) {
      float bv = bias[n0 + 16 * t + lo];
#pragma unroll
      for (int r = 0; r < 8; ++r) {
        float v = acc[t][r] + bv;
        if (RELU) v = fmaxf(v, 0.0f);
        long idx = (long)(m0 + r + (h << 3)) * Nn + n0 + 16 * t + lo;
        if constexpr (sizeof(OutT) == 2) C[idx] = f2bf(v);
        else                             C[idx] = v;
      }
    }
  }
}

static inline long cdiv(long a, long b) { return (a + b - 1) / b; }

extern "C" void kernel_launch(void* const* d_in, const int* in_sizes, int n_in,
                              void* d_out, int out_size, void* d_ws, size_t ws_size,
                              hipStream_t stream) {
  const float* x_game  = (const float*)d_in[0];
  const float* x_state = (const float*)d_in[1];
  const int* ei_gg   = (const int*)d_in[2];
  const int* ei_ss   = (const int*)d_in[3];
  const int* ei_hist = (const int*)d_in[4];
  const int* ei_in   = (const int*)d_in[5];
  const float* W1l = (const float*)d_in[6];
  const float* b1  = (const float*)d_in[7];
  const float* W1r = (const float*)d_in[8];
  const float* W2l = (const float*)d_in[9];
  const float* b2  = (const float*)d_in[10];
  const float* W2r = (const float*)d_in[11];
  const float* W3l = (const float*)d_in[12];
  const float* b3  = (const float*)d_in[13];
  const float* W3r = (const float*)d_in[14];
  const float* W4l = (const float*)d_in[15];
  const float* b4  = (const float*)d_in[16];
  const float* W4r = (const float*)d_in[17];
  const float* Wout = (const float*)d_in[18];
  const float* bout = (const float*)d_in[19];
  float* out = (float*)d_out;

  const int E = in_sizes[2] / 2;

  // ---- workspace carve-up (all 16B aligned) ----
  char* ws = (char*)d_ws;
  size_t off = 0;
  float*  agg   = (float*)(ws + off);  off += (size_t)NS_ * H_ * 4;        // 51.2 MB
  float*  cnt   = (float*)(ws + off);  off += (size_t)NS_ * 4;             // 200 KB
  __bf16* xg_bf = (__bf16*)(ws + off); off += (size_t)NG_ * DG_ * 2;       // 12.8 MB
  __bf16* xs_bf = (__bf16*)(ws + off); off += (size_t)NS_ * DG_ * 2;
  __bf16* g_bf  = (__bf16*)(ws + off); off += (size_t)NG_ * H_ * 2;        // 25.6 MB
  __bf16* sA_bf = (__bf16*)(ws + off); off += (size_t)NS_ * H_ * 2;
  __bf16* sB_bf = (__bf16*)(ws + off); off += (size_t)NS_ * H_ * 2;
  __bf16* Apack = (__bf16*)(ws + off); off += (size_t)NS_ * (2 * H_) * 2;  // 51.2 MB
  __bf16* wt1 = (__bf16*)(ws + off); off += (size_t)H_ * (2 * DG_) * 2;
  __bf16* wt2 = (__bf16*)(ws + off); off += (size_t)H_ * (2 * DG_) * 2;
  __bf16* wt3 = (__bf16*)(ws + off); off += (size_t)H_ * (2 * H_) * 2;
  __bf16* wt4 = (__bf16*)(ws + off); off += (size_t)H_ * (2 * H_) * 2;
  __bf16* wtO = (__bf16*)(ws + off); off += (size_t)OUT_ * H_ * 2;
  __bf16* s4_bf = sA_bf;  // sA dead after conv3 pack; reuse for conv4 output

  const dim3 blk(256, 1, 1);
  const dim3 gemm_grid((unsigned)cdiv(NS_ / 16, 8), H_ / 64, 1);   // 391 x 4
  const dim3 fin_grid((unsigned)cdiv(NS_ / 16, 8), OUT_ / 64, 1);  // 391 x 1

  // ---- input feature + weight conversion (tiny / one-shot) ----
  cvt_f32_bf16<<<2048, blk, 0, stream>>>(xg_bf, x_game, (long)NG_ * DG_);
  cvt_f32_bf16<<<2048, blk, 0, stream>>>(xs_bf, x_state, (long)NS_ * DG_);
  wprep<<<cdiv(H_ * 2 * DG_, 256), blk, 0, stream>>>(wt1, W1l, W1r, DG_, DG_, H_);
  wprep<<<cdiv(H_ * 2 * DG_, 256), blk, 0, stream>>>(wt2, W2l, W2r, DG_, DG_, H_);
  wprep<<<cdiv(H_ * 2 * H_, 256), blk, 0, stream>>>(wt3, W3l, W3r, H_, H_, H_);
  wprep<<<cdiv(H_ * 2 * H_, 256), blk, 0, stream>>>(wt4, W4l, W4r, H_, H_, H_);
  wprep<<<cdiv(OUT_ * H_, 256), blk, 0, stream>>>(wtO, Wout, Wout, H_, 0, OUT_);

  // ---- conv1: game -> g_bf  (d=128, K=256) ----
  zero_f32<<<4096, blk, 0, stream>>>(agg, (long)NG_ * DG_);
  zero_f32<<<256, blk, 0, stream>>>(cnt, NG_);
  scatter_add_bf<<<cdiv((long)E * (DG_ / 8), 256), blk, 0, stream>>>(
      xg_bf, ei_gg, agg, cnt, E, DG_);
  pack_a<<<cdiv((long)NG_ * 2 * DG_, 256), blk, 0, stream>>>(
      Apack, agg, cnt, xg_bf, NG_, DG_);
  wmma_gemm<2 * DG_, true, __bf16><<<gemm_grid, blk, 0, stream>>>(
      Apack, wt1, b1, g_bf, NG_, H_);

  // ---- conv2: state -> sA_bf ----
  zero_f32<<<4096, blk, 0, stream>>>(agg, (long)NS_ * DG_);
  zero_f32<<<256, blk, 0, stream>>>(cnt, NS_);
  scatter_add_bf<<<cdiv((long)E * (DG_ / 8), 256), blk, 0, stream>>>(
      xs_bf, ei_ss, agg, cnt, E, DG_);
  pack_a<<<cdiv((long)NS_ * 2 * DG_, 256), blk, 0, stream>>>(
      Apack, agg, cnt, xs_bf, NS_, DG_);
  wmma_gemm<2 * DG_, true, __bf16><<<gemm_grid, blk, 0, stream>>>(
      Apack, wt2, b2, sA_bf, NS_, H_);

  // ---- conv3: (g, sA) -> sB_bf  (d=256, K=512) ----
  zero_f32<<<4096, blk, 0, stream>>>(agg, (long)NS_ * H_);
  zero_f32<<<256, blk, 0, stream>>>(cnt, NS_);
  scatter_add_bf<<<cdiv((long)E * (H_ / 8), 256), blk, 0, stream>>>(
      g_bf, ei_hist, agg, cnt, E, H_);
  pack_a<<<cdiv((long)NS_ * 2 * H_, 256), blk, 0, stream>>>(
      Apack, agg, cnt, sA_bf, NS_, H_);
  wmma_gemm<2 * H_, true, __bf16><<<gemm_grid, blk, 0, stream>>>(
      Apack, wt3, b3, sB_bf, NS_, H_);

  // ---- conv4: (g, sB) -> s4_bf ----
  zero_f32<<<4096, blk, 0, stream>>>(agg, (long)NS_ * H_);
  zero_f32<<<256, blk, 0, stream>>>(cnt, NS_);
  scatter_add_bf<<<cdiv((long)E * (H_ / 8), 256), blk, 0, stream>>>(
      g_bf, ei_in, agg, cnt, E, H_);
  pack_a<<<cdiv((long)NS_ * 2 * H_, 256), blk, 0, stream>>>(
      Apack, agg, cnt, sB_bf, NS_, H_);
  wmma_gemm<2 * H_, true, __bf16><<<gemm_grid, blk, 0, stream>>>(
      Apack, wt4, b4, s4_bf, NS_, H_);

  // ---- final linear: s4 @ Wout + bout  (Nn=64, f32 out, no relu) ----
  wmma_gemm<H_, false, float><<<fin_grid, blk, 0, stream>>>(
      s4_bf, wtO, bout, out, NS_, OUT_);
}
